// ResNet_cifar10_48412871361036
// MI455X (gfx1250) — 16-bit and 8-bit matrix operand order repaired and checked by reading, compiled, not run on hardware
//
#include <hip/hip_runtime.h>
#include <hip/hip_bf16.h>
#include <stdint.h>

typedef __attribute__((ext_vector_type(4)))  int      v4i;
typedef __attribute__((ext_vector_type(8)))  int      v8i;
typedef __attribute__((ext_vector_type(8)))  float    v8f;
typedef __attribute__((ext_vector_type(16))) _Float16 v16h;

#if defined(__has_builtin)
#if __has_builtin(__builtin_amdgcn_global_load_async_to_lds_b128)
#define HAVE_ASYNC_LDS 1
#endif
#endif
#ifndef HAVE_ASYNC_LDS
#define HAVE_ASYNC_LDS 0
#endif

#if HAVE_ASYNC_LDS
static __device__ __forceinline__ void async_wait0() {
#if __has_builtin(__builtin_amdgcn_s_wait_asynccnt)
  __builtin_amdgcn_s_wait_asynccnt(0);
#else
  asm volatile("s_wait_asynccnt 0x0" ::: "memory");
#endif
}
#endif

__global__ void k_zero(float* p, int n) {
  int i = blockIdx.x * blockDim.x + threadIdx.x;
  if (i < n) p[i] = 0.f;
}

__global__ void k_binarize(const float* __restrict__ x, signed char* __restrict__ y, size_t n) {
  size_t stride = (size_t)gridDim.x * blockDim.x;
  for (size_t i = (size_t)blockIdx.x * blockDim.x + threadIdx.x; i < n; i += stride)
    y[i] = (x[i] >= 0.f) ? (signed char)1 : (signed char)-1;
}

__global__ void k_relu(float* x, size_t n) {
  size_t stride = (size_t)gridDim.x * blockDim.x;
  for (size_t i = (size_t)blockIdx.x * blockDim.x + threadIdx.x; i < n; i += stride)
    x[i] = fmaxf(x[i], 0.f);
}

__global__ void k_prelu(float* x, const float* __restrict__ a, size_t n) {
  float al = a[0];
  size_t stride = (size_t)gridDim.x * blockDim.x;
  for (size_t i = (size_t)blockIdx.x * blockDim.x + threadIdx.x; i < n; i += stride) {
    float v = x[i];
    x[i] = (v > 0.f) ? v : al * v;
  }
}

__global__ void k_add(float* x, const float* __restrict__ r, size_t n) {
  size_t stride = (size_t)gridDim.x * blockDim.x;
  for (size_t i = (size_t)blockIdx.x * blockDim.x + threadIdx.x; i < n; i += stride)
    x[i] += r[i];
}

__global__ void k_bias(float* x, const float* __restrict__ b, int C, size_t n) {
  size_t stride = (size_t)gridDim.x * blockDim.x;
  for (size_t i = (size_t)blockIdx.x * blockDim.x + threadIdx.x; i < n; i += stride)
    x[i] += b[(int)(i % (size_t)C)];
}

__global__ void k_avgpool(const float* __restrict__ x, float* __restrict__ y,
                          int Bn, int HW, int C) {
  int i = blockIdx.x * blockDim.x + threadIdx.x;
  if (i >= Bn * C) return;
  int bb = i / C, c = i - bb * C;
  float s = 0.f;
  for (int p = 0; p < HW; ++p) s += x[((size_t)bb * HW + p) * C + c];
  y[i] = s / (float)HW;
}

__global__ void k_bn_stats(const float* __restrict__ x, float* __restrict__ acc,
                           int C, size_t n) {
  __shared__ float ssum[320];
  __shared__ float ssq[320];
  for (int c = threadIdx.x; c < C; c += blockDim.x) { ssum[c] = 0.f; ssq[c] = 0.f; }
  __syncthreads();
  size_t stride = (size_t)gridDim.x * blockDim.x;
  for (size_t i = (size_t)blockIdx.x * blockDim.x + threadIdx.x; i < n; i += stride) {
    float v = x[i];
    int c = (int)(i % (size_t)C);
    atomicAdd(&ssum[c], v);
    atomicAdd(&ssq[c], v * v);
  }
  __syncthreads();
  for (int c = threadIdx.x; c < C; c += blockDim.x) {
    atomicAdd(&acc[c], ssum[c]);
    atomicAdd(&acc[C + c], ssq[c]);
  }
}

__global__ void k_bn_apply(float* x, const float* __restrict__ acc,
                           const float* __restrict__ g, const float* __restrict__ b,
                           int C, size_t n, float invcnt) {
  size_t stride = (size_t)gridDim.x * blockDim.x;
  for (size_t i = (size_t)blockIdx.x * blockDim.x + threadIdx.x; i < n; i += stride) {
    int c = (int)(i % (size_t)C);
    float m = acc[c] * invcnt;
    float var = acc[C + c] * invcnt - m * m;
    x[i] = (x[i] - m) * rsqrtf(var + 1e-5f) * g[c] + b[c];
  }
}

__global__ void k_pack_iu8(const float* __restrict__ w, signed char* __restrict__ wp,
                           int Cout, int Cin, int ks, int Kp, int Np) {
  int total = Np * Kp;
  int gs = gridDim.x * blockDim.x;
  int K = ks * ks * Cin;
  for (int i = blockIdx.x * blockDim.x + threadIdx.x; i < total; i += gs) {
    int n = i / Kp, k = i - n * Kp;
    signed char v = 0;
    if (n < Cout && k < K) {
      int tap = k / Cin, ch = k - tap * Cin;
      int ky = tap / ks, kx = tap - ky * ks;
      float wv = w[(((size_t)n * Cin + ch) * ks + ky) * ks + kx];
      v = (wv >= 0.f) ? (signed char)1 : (signed char)-1;
    }
    wp[i] = v;
  }
}

__global__ void k_pack_f16(const float* __restrict__ w, _Float16* __restrict__ wp,
                           int Cout, int Cin, int ks, int Kp, int Np) {
  int total = Np * Kp;
  int gs = gridDim.x * blockDim.x;
  int K = ks * ks * Cin;
  for (int i = blockIdx.x * blockDim.x + threadIdx.x; i < total; i += gs) {
    int n = i / Kp, k = i - n * Kp;
    _Float16 v = (_Float16)0.f;
    if (n < Cout && k < K) {
      int tap = k / Cin, ch = k - tap * Cin;
      int ky = tap / ks, kx = tap - ky * ks;
      float wv = w[(((size_t)n * Cin + ch) * ks + ky) * ks + kx];
      v = (wv >= 0.f) ? (_Float16)1.f : (_Float16)-1.f;
    }
    wp[i] = v;
  }
}

__global__ void __launch_bounds__(256)
k_bconv_iu8(const signed char* __restrict__ act,
            const signed char* __restrict__ wp,
            float* __restrict__ out,
            int Bn, int H, int W, int Cin,
            int Ho, int Wo, int Cout,
            int ks, int stride, int pad,
            int K, int Kp, int tilesN) {
  __shared__ signed char sB[1024];
  int waveId = threadIdx.x >> 5;
  int lane = threadIdx.x & 31;
  int tn = blockIdx.x % tilesN;
  int tmg = blockIdx.x / tilesN;
  int M = Bn * Ho * Wo;
  int tilesM = (M + 15) >> 4;
  int tm = tmg * 8 + waveId;
  bool tv = (tm < tilesM);
  int m0 = tm << 4, n0 = tn << 4;
  int r = lane & 15, half = lane >> 4;
  int m = m0 + r;
  bool mv = tv && (m < M);

  int ox = 0, oy = 0, bimg = 0;
  if (mv) { ox = m % Wo; int t = m / Wo; oy = t % Ho; bimg = t / Ho; }
  const signed char* abase = act + (size_t)bimg * H * W * Cin;
  int oyS = oy * stride - pad;
  int oxS = ox * stride - pad;

  int chs[4], kys[4], kxs[4];
#pragma unroll
  for (int c = 0; c < 4; ++c) { chs[c] = (c >> 1) * 32 + half * 16 + (c & 1) * 8; kys[c] = 0; kxs[c] = 0; }

  v8i acc = {};
  for (int k0 = 0; k0 < Kp; k0 += 64) {
    {
      int t = threadIdx.x;
      if (t < 64) {
        int col = t >> 2, seg = t & 3;
        const signed char* g = wp + (size_t)(n0 + col) * Kp + k0 + seg * 16;
#if HAVE_ASYNC_LDS
        signed char* gnc = const_cast<signed char*>(g);
        __builtin_amdgcn_global_load_async_to_lds_b128(
            (__attribute__((address_space(1))) v4i*)gnc,
            (__attribute__((address_space(3))) v4i*)(sB + t * 16), 0, 0);
        async_wait0();
#else
        int4 v = *(const int4*)g;
        *(int4*)(sB + t * 16) = v;
#endif
      }
      __syncthreads();
    }

    v8i b;
    {
      const signed char* bcol = sB + (lane & 15) * 64;
      int4 b0 = *(const int4*)(bcol + half * 16);
      int4 b1 = *(const int4*)(bcol + 32 + half * 16);
      b[0] = b0.x; b[1] = b0.y; b[2] = b0.z; b[3] = b0.w;
      b[4] = b1.x; b[5] = b1.y; b[6] = b1.z; b[7] = b1.w;
    }

    v8i a;
#pragma unroll
    for (int c = 0; c < 4; ++c) {
      int ky = kys[c], kx = kxs[c], ch = chs[c];
      int lo = 0, hi = 0;
      if (mv && ky < ks) {
        int iy = oyS + ky, ix = oxS + kx;
        if (iy >= 0 && iy < H && ix >= 0 && ix < W) {
          const int* p = (const int*)(abase + ((size_t)iy * W + ix) * Cin + ch);
          lo = p[0]; hi = p[1];
        }
      }
      a[2 * c] = lo; a[2 * c + 1] = hi;
      ch += 64;
      if (ch >= Cin) {
        ch -= Cin;
        if (++kx == ks) { kx = 0; ++ky; }
      }
      chs[c] = ch; kxs[c] = kx; kys[c] = ky;
    }

    acc = __builtin_amdgcn_wmma_i32_16x16x64_iu8(true, a, true, b, acc, false, false);
    __syncthreads();
  }

  if (tv) {
#pragma unroll
    for (int i = 0; i < 8; ++i) {
      int mm = m0 + i + 8 * half;
      int nn = n0 + (lane & 15);
      if (mm < M && nn < Cout)
        out[(size_t)mm * Cout + nn] = (float)acc[i];
    }
  }
}

__global__ void k_stem_f16(const float* __restrict__ x, const _Float16* __restrict__ wp,
                           float* __restrict__ out,
                           int Bn, int H, int W, int Cout, int tilesN) {
  int wave = blockIdx.x * (blockDim.x >> 5) + (threadIdx.x >> 5);
  int lane = threadIdx.x & 31;
  int M = Bn * H * W;
  int tilesM = (M + 15) >> 4;
  if (wave >= tilesM * tilesN) return;
  int tm = wave / tilesN, tn = wave - tm * tilesN;
  int m0 = tm << 4, n0 = tn << 4;
  int r = lane & 15, half = lane >> 4;
  int m = m0 + r;
  bool mv = (m < M);
  int ox = 0, oy = 0, bimg = 0;
  if (mv) { ox = m % W; int t = m / W; oy = t % H; bimg = t / H; }

  v16h a;
#pragma unroll
  for (int j = 0; j < 16; ++j) {
    int k = half * 16 + j;
    _Float16 val = (_Float16)0.f;
    if (mv && k < 27) {
      int tap = k / 3, ch = k - tap * 3;
      int ky = tap / 3, kx = tap - ky * 3;
      int iy = oy + ky - 1, ix = ox + kx - 1;
      if (iy >= 0 && iy < H && ix >= 0 && ix < W)
        val = (_Float16)x[(((size_t)bimg * 3 + ch) * H + iy) * W + ix];
    }
    a[j] = val;
  }
  v16h b;
  {
    const _Float16* br = wp + (size_t)(n0 + (lane & 15)) * 32 + half * 16;
#pragma unroll
    for (int j = 0; j < 16; ++j) b[j] = br[j];
  }
  v8f c = {};
  c = __builtin_amdgcn_wmma_f32_16x16x32_f16(false, a, false, b, (short)0, c, false, false);
#pragma unroll
  for (int i = 0; i < 8; ++i) {
    int mm = m0 + i + 8 * half;
    int nn = n0 + (lane & 15);
    if (mm < M && nn < Cout)
      out[(size_t)mm * Cout + nn] = c[i];
  }
}

extern "C" void kernel_launch(void* const* d_in, const int* in_sizes, int n_in,
                              void* d_out, int out_size, void* d_ws, size_t ws_size,
                              hipStream_t stream) {
  (void)in_sizes; (void)n_in; (void)out_size; (void)ws_size;
  auto F = [&](int i) { return (const float*)d_in[i]; };
  const float* x      = F(0);
  const float* stem_w = F(1);
  const float* stem_g = F(2);
  const float* stem_b = F(3);

  struct Blk { const float *w1, *w2, *g1, *b1, *g2, *b2, *pre, *wd, *gd, *bd; };
  Blk blks[6];
  const bool hasds[6] = {false, false, true, false, true, false};
  int idx = 4;
  for (int b = 0; b < 6; ++b) {
    Blk& k = blks[b];
    k.w1 = F(idx++); k.w2 = F(idx++);
    k.g1 = F(idx++); k.b1 = F(idx++);
    k.g2 = F(idx++); k.b2 = F(idx++);
    k.pre = F(idx++);
    if (hasds[b]) { k.wd = F(idx++); k.gd = F(idx++); k.bd = F(idx++); }
    else          { k.wd = nullptr;  k.gd = nullptr;  k.bd = nullptr; }
  }
  const float* fc_w  = F(idx++);
  const float* fc_b  = F(idx++);
  const float* bn3_g = F(idx++);
  const float* bn3_b = F(idx++);

  char* ws = (char*)d_ws;
  size_t off = 0;
  auto carve = [&](size_t bytes) -> char* {
    char* p = ws + off;
    off += (bytes + 255) & ~(size_t)255;
    return p;
  };
  const size_t ACT = (size_t)256 * 32 * 32 * 80;
  float*       bufA   = (float*)carve(ACT * 4);
  float*       bufB   = (float*)carve(ACT * 4);
  float*       bufC   = (float*)carve(ACT * 4);
  signed char* bin    = (signed char*)carve(ACT);
  signed char* wpack  = (signed char*)carve((size_t)320 * 2880);
  _Float16*    wp16   = (_Float16*)carve((size_t)80 * 32 * 2);
  float*       accum  = (float*)carve((size_t)2 * 320 * 4);
  float*       pooled = (float*)carve((size_t)256 * 320 * 4);
  signed char* pbin   = (signed char*)carve((size_t)256 * 320);

  auto eg = [](size_t n) -> unsigned {
    size_t g = (n + 255) / 256;
    return (unsigned)(g > 16384 ? 16384 : (g == 0 ? 1 : g));
  };

  auto bconv = [&](const signed char* act, const float* w, float* outp,
                   int Bn, int H, int W, int Cin, int Cout, int ks, int stride, int pad) {
    int Ho = (H + 2 * pad - ks) / stride + 1;
    int Wo = (W + 2 * pad - ks) / stride + 1;
    int K  = ks * ks * Cin;
    int Kp = (K + 63) & ~63;
    int Np = (Cout + 15) & ~15;
    k_pack_iu8<<<eg((size_t)Np * Kp), 256, 0, stream>>>(w, wpack, Cout, Cin, ks, Kp, Np);
    int M = Bn * Ho * Wo;
    int tilesM = (M + 15) >> 4, tilesN = Np >> 4;
    int tmg = (tilesM + 7) / 8;
    k_bconv_iu8<<<tmg * tilesN, 256, 0, stream>>>(act, wpack, outp,
        Bn, H, W, Cin, Ho, Wo, Cout, ks, stride, pad, K, Kp, tilesN);
  };

  auto bn = [&](float* buf, const float* g, const float* b, int C, size_t total, float cnt) {
    k_zero<<<(2 * C + 255) / 256, 256, 0, stream>>>(accum, 2 * C);
    k_bn_stats<<<eg(total), 256, 0, stream>>>(buf, accum, C, total);
    k_bn_apply<<<eg(total), 256, 0, stream>>>(buf, accum, g, b, C, total, 1.f / cnt);
  };

  k_pack_f16<<<eg((size_t)80 * 32), 256, 0, stream>>>(stem_w, wp16, 80, 3, 3, 32, 80);
  {
    int M = 256 * 32 * 32;
    int tilesM = (M + 15) >> 4, tilesN = 80 / 16;
    int waves = tilesM * tilesN;
    k_stem_f16<<<(waves + 7) / 8, 256, 0, stream>>>(x, wp16, bufA, 256, 32, 32, 80, tilesN);
  }
  size_t nStem = (size_t)256 * 32 * 32 * 80;
  k_relu<<<eg(nStem), 256, 0, stream>>>(bufA, nStem);
  bn(bufA, stem_g, stem_b, 80, nStem, 256.f * 32.f * 32.f);

  auto block = [&](float* X, float* T1, float* T2, const Blk& p,
                   int Hin, int Win, int Cin, int Cout, int stride) {
    int Ho = Hin / stride, Wo = Win / stride;
    size_t nIn  = (size_t)256 * Hin * Win * Cin;
    size_t nOut = (size_t)256 * Ho * Wo * Cout;
    float cnt = 256.f * (float)Ho * (float)Wo;
    k_binarize<<<eg(nIn), 256, 0, stream>>>(X, bin, nIn);
    bconv(bin, p.w1, T1, 256, Hin, Win, Cin, Cout, 3, stride, 1);
    k_prelu<<<eg(nOut), 256, 0, stream>>>(T1, p.pre, nOut);
    bn(T1, p.g1, p.b1, Cout, nOut, cnt);
    k_binarize<<<eg(nOut), 256, 0, stream>>>(T1, bin, nOut);
    bconv(bin, p.w2, T2, 256, Ho, Wo, Cout, Cout, 3, 1, 1);
    k_prelu<<<eg(nOut), 256, 0, stream>>>(T2, p.pre, nOut);
    bn(T2, p.g2, p.b2, Cout, nOut, cnt);
    if (p.wd) {
      k_binarize<<<eg(nIn), 256, 0, stream>>>(X, bin, nIn);
      bconv(bin, p.wd, T1, 256, Hin, Win, Cin, Cout, 1, stride, 0);
      bn(T1, p.gd, p.bd, Cout, nOut, cnt);
      k_add<<<eg(nOut), 256, 0, stream>>>(T2, T1, nOut);
    } else {
      k_add<<<eg(nOut), 256, 0, stream>>>(T2, X, nOut);
    }
  };

  block(bufA, bufB, bufC, blks[0], 32, 32,  80,  80, 1);
  block(bufC, bufB, bufA, blks[1], 32, 32,  80,  80, 1);
  block(bufA, bufB, bufC, blks[2], 32, 32,  80, 160, 2);
  block(bufC, bufB, bufA, blks[3], 16, 16, 160, 160, 1);
  block(bufA, bufB, bufC, blks[4], 16, 16, 160, 320, 2);
  block(bufC, bufB, bufA, blks[5],  8,  8, 320, 320, 1);

  size_t nFin = (size_t)256 * 8 * 8 * 320;
  k_relu<<<eg(nFin), 256, 0, stream>>>(bufA, nFin);
  k_avgpool<<<(256 * 320 + 255) / 256, 256, 0, stream>>>(bufA, pooled, 256, 64, 320);
  k_binarize<<<eg((size_t)256 * 320), 256, 0, stream>>>(pooled, pbin, (size_t)256 * 320);

  float* logits = (float*)d_out;
  bconv(pbin, fc_w, logits, 256, 1, 1, 320, 10, 1, 1, 0);
  k_bias<<<(2560 + 255) / 256, 256, 0, stream>>>(logits, fc_b, 10, (size_t)2560);
  bn(logits, bn3_g, bn3_b, 10, (size_t)2560, 256.f);
}
